// RNN_GRU_Naive_Contextual_45251775430855
// MI455X (gfx1250) — matrix operand order repaired and checked by reading, not compiled, not run on hardware
//
#include <hip/hip_runtime.h>
#include <hip/hip_bf16.h>

#define B_   256
#define T_   1024
#define IN_  64
#define EMB_ 32
#define H_   256
#define G3   (3 * H_)   // 768

typedef __attribute__((ext_vector_type(16))) __bf16 v16bf;
typedef __attribute__((ext_vector_type(8)))  float  v8f;

union BF16Frag { v16bf v; int4 q[2]; };

__device__ __forceinline__ float sigmoidf_(float x) {
    return 1.0f / (1.0f + __expf(-x));
}

#define WMMA_BF16(acc, a, b) \
    acc = __builtin_amdgcn_wmma_f32_16x16x32_bf16(false, (a), false, (b), (short)0, (acc), false, false)

// ---------------- weight conversion: f32 -> bf16 into workspace ----------------
__global__ void __launch_bounds__(256)
gru_convert_weights(const float* __restrict__ Whh,
                    const float* __restrict__ Wih,
                    __bf16* __restrict__ out) {
    int i = blockIdx.x * blockDim.x + threadIdx.x;
    if (i < G3 * H_)   out[i]           = (__bf16)Whh[i];
    if (i < G3 * EMB_) out[G3 * H_ + i] = (__bf16)Wih[i];
}

// ---------------- fused GRU scan: 16 blocks x 16 batch rows, 8 waves/block ----------------
// Wave w owns gate tiles {2w,2w+1} (r), {16+2w,16+2w+1} (z), {32+2w,33+2w} (n):
// identical lane/element mapping for all three gates of hidden cols j in [32w,32w+32),
// so gating and the fp32 hidden state are fully register-resident (output-stationary).
__global__ void __launch_bounds__(256, 1)
gru_scan_kernel(const float*  __restrict__ x,
                const float*  __restrict__ W_lin,
                const float*  __restrict__ b_lin,
                const float*  __restrict__ b_ih,
                const float*  __restrict__ b_hh,
                const __bf16* __restrict__ Wih_bf,   // [768][32]
                const __bf16* __restrict__ Whh_bf,   // [768][256]
                const float*  __restrict__ W_out,
                const float*  __restrict__ b_out,
                float*        __restrict__ out) {
    __shared__ __bf16 h_bf [16 * H_];     //  8 KB : bf16 h (WMMA A operand, all-K shared)
    __shared__ float  x_lds[16 * IN_];    //  4 KB : staged x tile
    __shared__ __bf16 xe_bf[16 * EMB_];   //  1 KB : embedded input (A operand)
    __shared__ float  wlin_lds[EMB_ * IN_]; // 8 KB : W_lin (avoid per-step global reads)
    __shared__ float  hfin [16 * H_];     // 16 KB : final h (projection only)

    const int tid  = threadIdx.x;
    const int lane = tid & 31;
    const int w    = tid >> 5;     // wave 0..7
    const int n    = lane & 15;    // col within 16x16 tile
    const int kh   = lane >> 4;    // K-half selector
    const int bg   = blockIdx.x * 16;

    for (int i = tid; i < 16 * H_; i += 256)  h_bf[i] = (__bf16)0.0f;
    for (int i = tid; i < EMB_ * IN_; i += 256) wlin_lds[i] = W_lin[i];
    __syncthreads();

    // staging / xe-compute thread mappings
    const int xr  = tid >> 4;        // row for x staging
    const int xc  = (tid & 15) * 4;  // float4 column for x staging
    const int er  = tid & 15;        // row for xe
    const int ec0 = tid >> 4;        // first of two xe columns

    // fragment loaders (ISA 16-bit A / B VGPR layouts, two b128 loads each)
    auto loadA = [&](const __bf16* sp, int kb, int stride) -> v16bf {
        BF16Frag a;
        const __bf16* p = sp + n * stride + kb + kh * 8;
        a.q[0] = *reinterpret_cast<const int4*>(p);        // K = kb+kh*8   .. +7
        a.q[1] = *reinterpret_cast<const int4*>(p + 16);   // K = kb+16+kh*8.. +7
        return a.v;
    };
    auto loadB = [&](const __bf16* Wp, int gbase, int kb, int stride) -> v16bf {
        BF16Frag b;
        const __bf16* p = Wp + (size_t)(gbase + n) * stride + kb + kh * 8;
        b.q[0] = *reinterpret_cast<const int4*>(p);        // K = kb+kh*8   .. +7
        b.q[1] = *reinterpret_cast<const int4*>(p + 16);   // K = kb+16+kh*8.. +7
        return b.v;
    };

    // per-lane hidden-column ownership: cols c0/c1 for the two owned tiles
    const int c0 = (2 * w + 0) * 16 + n;   // j for q=0
    const int c1 = (2 * w + 1) * 16 + n;   // j for q=1

    // t-invariant biases, register-resident (8 scalars)
    const float bR0 = b_ih[c0]       + b_hh[c0];
    const float bR1 = b_ih[c1]       + b_hh[c1];
    const float bZ0 = b_ih[256 + c0] + b_hh[256 + c0];
    const float bZ1 = b_ih[256 + c1] + b_hh[256 + c1];
    const float bX0 = b_ih[512 + c0], bX1 = b_ih[512 + c1];
    const float bH0 = b_hh[512 + c0], bH1 = b_hh[512 + c1];

    // t-invariant W_ih B-fragments, deliberately register-resident (6 x 8 VGPRs)
    const v16bf wiR0 = loadB(Wih_bf, (2 * w + 0) * 16,       0, EMB_);
    const v16bf wiR1 = loadB(Wih_bf, (2 * w + 1) * 16,       0, EMB_);
    const v16bf wiZ0 = loadB(Wih_bf, 256 + (2 * w + 0) * 16, 0, EMB_);
    const v16bf wiZ1 = loadB(Wih_bf, 256 + (2 * w + 1) * 16, 0, EMB_);
    const v16bf wiX0 = loadB(Wih_bf, 512 + (2 * w + 0) * 16, 0, EMB_);
    const v16bf wiX1 = loadB(Wih_bf, 512 + (2 * w + 1) * 16, 0, EMB_);

    v8f hF0 = {}, hF1 = {};   // fp32 carried hidden state (register-resident)

    #pragma unroll 1
    for (int t = 0; t < T_; ++t) {
        // Opaque zero regenerated every iteration: keeps W_hh streaming loads
        // inside the loop (prevents LICM hoisting ~400 VGPRs -> scratch spills).
        int zero_;
        asm volatile("s_mov_b32 %0, 0" : "=s"(zero_));
        const __bf16* Whh_t = Whh_bf + zero_;

        // ---- stage x[bg..bg+15, t, :] into LDS (coalesced float4) ----
        *reinterpret_cast<float4*>(&x_lds[xr * IN_ + xc]) =
            *reinterpret_cast<const float4*>(&x[((size_t)(bg + xr) * T_ + t) * IN_ + xc]);
        __syncthreads();

        // ---- xe = x_tile @ W_lin^T + b_lin  (16x32, VALU) ----
        #pragma unroll
        for (int p = 0; p < 2; ++p) {
            int c = ec0 + p * 16;
            float s = b_lin[c];
            const float4* xr4 = reinterpret_cast<const float4*>(&x_lds[er * IN_]);
            const float4* wr4 = reinterpret_cast<const float4*>(&wlin_lds[c * IN_]);
            #pragma unroll
            for (int i = 0; i < IN_ / 4; ++i) {
                float4 a4 = xr4[i], b4 = wr4[i];
                s += a4.x * b4.x + a4.y * b4.y + a4.z * b4.z + a4.w * b4.w;
            }
            xe_bf[er * EMB_ + c] = (__bf16)s;
        }
        __syncthreads();

        // ---- WMMA accumulation: gx (K=32) + gh (K=256) ----
        v8f aR0 = {}, aR1 = {}, aZ0 = {}, aZ1 = {};
        v8f aX0 = {}, aX1 = {}, aH0 = {}, aH1 = {};

        {   // input-gate contribution from register-resident W_ih fragments
            v16bf ax = loadA(xe_bf, 0, EMB_);
            WMMA_BF16(aR0, ax, wiR0);
            WMMA_BF16(aR1, ax, wiR1);
            WMMA_BF16(aZ0, ax, wiZ0);
            WMMA_BF16(aZ1, ax, wiZ1);
            WMMA_BF16(aX0, ax, wiX0);
            WMMA_BF16(aX1, ax, wiX1);
        }
        #pragma unroll
        for (int kc = 0; kc < 8; ++kc) {   // hidden contribution, W_hh streamed from L2
            int kb = kc * 32;
            v16bf ah = loadA(h_bf, kb, H_);
            WMMA_BF16(aR0, ah, loadB(Whh_t, (2 * w + 0) * 16,       kb, H_));
            WMMA_BF16(aR1, ah, loadB(Whh_t, (2 * w + 1) * 16,       kb, H_));
            WMMA_BF16(aZ0, ah, loadB(Whh_t, 256 + (2 * w + 0) * 16, kb, H_));
            WMMA_BF16(aZ1, ah, loadB(Whh_t, 256 + (2 * w + 1) * 16, kb, H_));
            WMMA_BF16(aH0, ah, loadB(Whh_t, 512 + (2 * w + 0) * 16, kb, H_));
            WMMA_BF16(aH1, ah, loadB(Whh_t, 512 + (2 * w + 1) * 16, kb, H_));
        }
        __syncthreads();   // all waves done reading h_bf of step t-1

        // ---- gates + state update, fully register-resident ----
        #pragma unroll
        for (int i = 0; i < 8; ++i) {
            int M = i + kh * 8;
            {
                float r  = sigmoidf_(aR0[i] + bR0);
                float z  = sigmoidf_(aZ0[i] + bZ0);
                float nn = tanhf(aX0[i] + bX0 + r * (aH0[i] + bH0));
                float hv = (1.0f - z) * nn + z * hF0[i];
                hF0[i] = hv;
                h_bf[M * H_ + c0] = (__bf16)hv;
            }
            {
                float r  = sigmoidf_(aR1[i] + bR1);
                float z  = sigmoidf_(aZ1[i] + bZ1);
                float nn = tanhf(aX1[i] + bX1 + r * (aH1[i] + bH1));
                float hv = (1.0f - z) * nn + z * hF1[i];
                hF1[i] = hv;
                h_bf[M * H_ + c1] = (__bf16)hv;
            }
        }
        // next iteration's post-x barrier separates these writes from the next reads
    }

    // ---- final projection: out = h_T @ W_out^T + b_out ----
    #pragma unroll
    for (int i = 0; i < 8; ++i) {
        int M = i + kh * 8;
        hfin[M * H_ + c0] = hF0[i];
        hfin[M * H_ + c1] = hF1[i];
    }
    __syncthreads();
    if (tid < 32) {
        int r = tid >> 1, o = tid & 1;
        float s = b_out[o];
        for (int j = 0; j < H_; ++j)
            s += hfin[r * H_ + j] * W_out[o * H_ + j];
        out[(bg + r) * 2 + o] = s;
    }
}

extern "C" void kernel_launch(void* const* d_in, const int* in_sizes, int n_in,
                              void* d_out, int out_size, void* d_ws, size_t ws_size,
                              hipStream_t stream) {
    const float* x     = (const float*)d_in[0];
    const float* W_lin = (const float*)d_in[1];
    const float* b_lin = (const float*)d_in[2];
    const float* W_ih  = (const float*)d_in[3];
    const float* b_ih  = (const float*)d_in[4];
    const float* W_hh  = (const float*)d_in[5];
    const float* b_hh  = (const float*)d_in[6];
    const float* W_out = (const float*)d_in[7];
    const float* b_out = (const float*)d_in[8];

    __bf16* wbf = (__bf16*)d_ws;   // [0,768*256): Whh_bf ; then 768*32: Wih_bf
    gru_convert_weights<<<(G3 * H_ + 255) / 256, 256, 0, stream>>>(W_hh, W_ih, wbf);
    gru_scan_kernel<<<B_ / 16, 256, 0, stream>>>(x, W_lin, b_lin, b_ih, b_hh,
                                                 wbf + (size_t)G3 * H_,  // Wih_bf
                                                 wbf,                    // Whh_bf
                                                 W_out, b_out, (float*)d_out);
}
